// STGAE_75814762709661
// MI455X (gfx1250) — matrix operand order repaired and checked by reading, not compiled, not run on hardware
//
#include <hip/hip_runtime.h>

// ---------------------------------------------------------------------------
// Shapes (compile-time constants from the reference)
// ---------------------------------------------------------------------------
#define Bsz 256
#define Nn  64
#define Tt  50
#define Ff  2
#define Ee  64
#define Hh  128
#define Rr  30
#define G4H 512   // 4*H

typedef __attribute__((ext_vector_type(16))) __bf16 v16bf;
typedef __attribute__((ext_vector_type(8)))  float  v8f;

// ---------------------------------------------------------------------------
// Helpers
// ---------------------------------------------------------------------------
__device__ __forceinline__ unsigned short f2bf(float f) {
    union { __bf16 b; unsigned short u; } cv;
    cv.b = (__bf16)f;                 // hardware v_cvt (RNE)
    return cv.u;
}
__device__ __forceinline__ float bf2f(unsigned short u) {
    union { unsigned int i; float f; } x;
    x.i = ((unsigned int)u) << 16;    // exact widening
    return x.f;
}

// gfx1250 has V_TANH_F32 (TRANS op, co-executes with VALU)
__device__ __forceinline__ float tanh_fast(float x) {
#if __has_builtin(__builtin_amdgcn_tanhf)
    return __builtin_amdgcn_tanhf(x);
#else
    return tanhf(x);
#endif
}
__device__ __forceinline__ float sig_fast(float x) {
    return 0.5f * tanh_fast(0.5f * x) + 0.5f;   // sigmoid via tanh identity
}

union Frag128 { v16bf v; uint4 q[2]; };

// Load a 16-bf16 fragment stored contiguously (32B) — works for LDS or global
__device__ __forceinline__ v16bf load_frag(const unsigned short* base) {
    Frag128 f;
    const uint4* p = reinterpret_cast<const uint4*>(base);
    f.q[0] = p[0];
    f.q[1] = p[1];
    return f.v;
}

// Inverse of the ISA 7.12.2 16-bit A-matrix layout:
// element (m, k) of a 16x32 block lives at lane = m + ((k&8)?16:0),
// e = (k&7) + ((k&31)>=16 ? 8 : 0), fragment slot = (lane*16 + e).
__device__ __forceinline__ int a_slot(int m, int r /* k&31 */) {
    int lane = m + ((r & 8) ? 16 : 0);
    int e    = (r & 7) + ((r >= 16) ? 8 : 0);
    return lane * 16 + e;
}

// ---------------------------------------------------------------------------
// Weight packing: W (K x 512, f32) -> bf16 wave32 B-fragment layout.
// flat = ((nb*KB + kb)*32 + lane)*16 + e
//   K = kb*32 + (lane>=16 ? 8 : 0) + (e&7) + (e>=8 ? 16 : 0) ;  N = nb*16 + lane%16
// ---------------------------------------------------------------------------
__global__ void pack_w_kernel(const float* __restrict__ W,
                              unsigned short* __restrict__ out, int KB) {
    const int total = 32 * KB * 32 * 16;
    for (int idx = blockIdx.x * blockDim.x + threadIdx.x; idx < total;
         idx += gridDim.x * blockDim.x) {
        int e    = idx & 15;
        int lane = (idx >> 4) & 31;
        int kb   = (idx >> 9) % KB;
        int nb   = idx / (512 * KB);
        int K    = kb * 32 + ((lane >= 16) ? 8 : 0) + (e & 7) + ((e >= 8) ? 16 : 0);
        int Ncol = nb * 16 + (lane & 15);
        out[idx] = f2bf(W[K * G4H + Ncol]);
    }
}

// ---------------------------------------------------------------------------
// Phase A: GCN front-end. Only node 0 of enc is consumed, and with F=2 the
// einsum collapses to two scalars s0,s1. Output is written DIRECTLY in the
// bf16 WMMA A-fragment layout consumed by the encoder.
// One block per (b,t), 64 threads.
// ---------------------------------------------------------------------------
__global__ __launch_bounds__(64) void stgae_gcn_kernel(
    const float* __restrict__ src, const float* __restrict__ Wres,
    const float* __restrict__ Wg, const float* __restrict__ bg,
    unsigned short* __restrict__ xpk) {
    __shared__ float xs[Nn * 2];
    __shared__ float dinv[Nn];
    __shared__ float cc0[Nn];
    __shared__ float cc1[Nn];

    const int bt = blockIdx.x;
    const int b  = bt / Tt;
    const int t  = bt % Tt;
    const int j  = threadIdx.x;

    // x[b,t,j,:] = src[b,j,t,:]
    xs[j * 2 + 0] = src[(((size_t)b * Nn + j) * Tt + t) * Ff + 0];
    xs[j * 2 + 1] = src[(((size_t)b * Nn + j) * Tt + t) * Ff + 1];
    __syncthreads();

    const float xj0 = xs[j * 2 + 0], xj1 = xs[j * 2 + 1];
    float rs = 0.0f;
    for (int k = 0; k < Nn; ++k) {
        float dx = xj0 - xs[k * 2 + 0];
        float dy = xj1 - xs[k * 2 + 1];
        float d  = sqrtf(dx * dx + dy * dy);
        // dist==0 -> 1e-6 -> 1/dist=1e6 -> min(.,1)=1 ; diagonal forced to 1
        float w = (k == j || d <= 0.0f) ? 1.0f : fminf(1.0f / d, 1.0f);
        rs += w;
    }
    dinv[j] = 1.0f / sqrtf(rs);

    float dx0 = xs[0] - xj0, dy0 = xs[1] - xj1;
    float d0  = sqrtf(dx0 * dx0 + dy0 * dy0);
    float w0  = (j == 0 || d0 <= 0.0f) ? 1.0f : fminf(1.0f / d0, 1.0f);
    __syncthreads();

    float a0 = w0 * dinv[0] * dinv[j];   // A[0][j]
    cc0[j]   = a0 * xj0;
    cc1[j]   = a0 * xj1;
    __syncthreads();

    float s0 = 0.0f, s1 = 0.0f;
    for (int k = 0; k < Nn; ++k) { s0 += cc0[k]; s1 += cc1[k]; }

    // emb[e=j] = relu(s0*Wg[0,j] + s1*Wg[1,j] + bg[j] + x0.Wres[:,j])
    float v = s0 * Wg[j] + s1 * Wg[Ee + j] + bg[j]
            + xs[0] * Wres[j] + xs[1] * Wres[Ee + j];
    v = fmaxf(v, 0.0f);

    // Scatter into the encoder's A-fragment layout (batch row m = b%16)
    const int m     = b & 15;
    const int btile = b >> 4;
    const int kb    = j >> 5;
    const int slot  = a_slot(m, j & 31);
    xpk[((((size_t)btile * Tt + t) * 2 + kb) * 32) * 16 + slot] = f2bf(v);
}

// ---------------------------------------------------------------------------
// One fused LSTM step, all-register gates.
// Wave w computes column tiles nb = {w, w+8, w+16, w+24}: the SAME 16 hidden
// columns of each gate (i,f,g,o). After 24 chained v_wmma_f32_16x16x32_bf16,
// a lane holds all four gate values for its (m, hcol) slots in registers, so
// activations + cell update run without any LDS round-trip; c is a persistent
// per-thread register array. h is published as bf16 in A-fragment layout to
// the double-buffered LDS mirror (read buf != write buf -> ONE barrier/step).
// ---------------------------------------------------------------------------
__device__ __forceinline__ void lstm_step_reg(
    v16bf ax0, v16bf ax1,
    const unsigned short* __restrict__ hbf_rd,
    unsigned short* __restrict__ hbf_wr,
    float* c_reg, const float* bb /* [4] hoisted biases */,
    const unsigned short* __restrict__ WihP,
    const unsigned short* __restrict__ WhhP) {
    const int tid  = threadIdx.x;
    const int lane = tid & 31;
    const int w    = tid >> 5;

    v16bf ah0 = load_frag(hbf_rd + (0 * 32 + lane) * 16);
    v16bf ah1 = load_frag(hbf_rd + (1 * 32 + lane) * 16);
    v16bf ah2 = load_frag(hbf_rd + (2 * 32 + lane) * 16);
    v16bf ah3 = load_frag(hbf_rd + (3 * 32 + lane) * 16);

    v8f acc[4];
#pragma unroll
    for (int g = 0; g < 4; ++g) {
        const int nb = w + 8 * g;
        const float b0 = bb[g];
        v8f a = {b0, b0, b0, b0, b0, b0, b0, b0};
        // X @ W_ih : K = 64 -> 2 fragments
        a = __builtin_amdgcn_wmma_f32_16x16x32_bf16(
            false, ax0, false, load_frag(WihP + ((nb * 2 + 0) * 32 + lane) * 16),
            (short)0, a, false, false);
        a = __builtin_amdgcn_wmma_f32_16x16x32_bf16(
            false, ax1, false, load_frag(WihP + ((nb * 2 + 1) * 32 + lane) * 16),
            (short)0, a, false, false);
        // H @ W_hh : K = 128 -> 4 fragments
        a = __builtin_amdgcn_wmma_f32_16x16x32_bf16(
            false, ah0, false, load_frag(WhhP + ((nb * 4 + 0) * 32 + lane) * 16),
            (short)0, a, false, false);
        a = __builtin_amdgcn_wmma_f32_16x16x32_bf16(
            false, ah1, false, load_frag(WhhP + ((nb * 4 + 1) * 32 + lane) * 16),
            (short)0, a, false, false);
        a = __builtin_amdgcn_wmma_f32_16x16x32_bf16(
            false, ah2, false, load_frag(WhhP + ((nb * 4 + 2) * 32 + lane) * 16),
            (short)0, a, false, false);
        a = __builtin_amdgcn_wmma_f32_16x16x32_bf16(
            false, ah3, false, load_frag(WhhP + ((nb * 4 + 3) * 32 + lane) * 16),
            (short)0, a, false, false);
        acc[g] = a;
    }

    // Register-resident gate fusion. This lane owns (m = v + mo, k = 16w+l15):
    // h scatter slot: kb = w>>1 ; slot lane = v + mo + ((lane&8)?16:0) ;
    // e = (lane&7) + ((w&1)?8:0)  (consecutive v -> +16 ushorts).
    const int mo  = (lane >= 16) ? 8 : 0;
    const int kb  = w >> 1;
    const int lhi = (lane & 8) ? 16 : 0;
    const int e   = (lane & 7) + ((w & 1) ? 8 : 0);
    unsigned short* hp = hbf_wr + (kb * 32 + mo + lhi) * 16 + e;
#pragma unroll
    for (int v = 0; v < 8; ++v) {
        float cn = sig_fast(acc[1][v]) * c_reg[v]
                 + sig_fast(acc[0][v]) * tanh_fast(acc[2][v]);
        float hn = sig_fast(acc[3][v]) * tanh_fast(cn);
        c_reg[v] = cn;
        hp[v * 16] = f2bf(hn);
    }
    __syncthreads();   // h published; next step may read the written buffer
}

// ---------------------------------------------------------------------------
// Phase B: encoder (50 steps) + decoder (30 steps), one block per 16 batch
// rows — recurrence is independent across batch, so no grid-wide sync.
// ---------------------------------------------------------------------------
__global__ __launch_bounds__(256) void stgae_lstm_kernel(
    const unsigned short* __restrict__ xpk, const float* __restrict__ src,
    const unsigned short* __restrict__ WihE,
    const unsigned short* __restrict__ WhhE, const float* __restrict__ bE,
    const float* __restrict__ W1, const float* __restrict__ b1,
    const unsigned short* __restrict__ WihD,
    const unsigned short* __restrict__ WhhD, const float* __restrict__ bD,
    const float* __restrict__ W2, const float* __restrict__ b2,
    float* __restrict__ out) {
    __shared__ alignas(16) unsigned short hbf[2][4 * 512]; // 2x4KB, h A-layout
    __shared__ alignas(16) unsigned short edbf[2 * 512];   // 2KB, ed A-layout
    __shared__ float xin[32];

    const int tid  = threadIdx.x;
    const int lane = tid & 31;
    const int w    = tid >> 5;
    const int l15  = lane & 15;
    const int b0   = blockIdx.x * 16;

    // Persistent per-thread cell state + hoisted biases
    float c_reg[8];
#pragma unroll
    for (int v = 0; v < 8; ++v) c_reg[v] = 0.0f;
    float bbE[4], bbD[4];
#pragma unroll
    for (int g = 0; g < 4; ++g) {
        bbE[g] = bE[(w + 8 * g) * 16 + l15];
        bbD[g] = bD[(w + 8 * g) * 16 + l15];
    }

    for (int i = tid; i < 4 * 512; i += 256) hbf[0][i] = 0;
    __syncthreads();

    int pr = 0;   // read buffer parity

    // ---- Encoder: x fragments come pre-packed from the GCN kernel ----
    for (int t = 0; t < Tt; ++t) {
        const unsigned short* xp =
            xpk + (((size_t)blockIdx.x * Tt + t) * 2) * 32 * 16;
        v16bf ax0 = load_frag(xp + lane * 16);
        v16bf ax1 = load_frag(xp + (32 + lane) * 16);
        lstm_step_reg(ax0, ax1, hbf[pr], hbf[pr ^ 1], c_reg, bbE, WihE, WhhE);
        pr ^= 1;
    }

    // ---- Decoder: xin = src[b, node0, T-1, :] ----
    if (tid < 32) {
        int m = tid >> 1, f = tid & 1;
        xin[tid] = src[(((size_t)(b0 + m) * Nn + 0) * Tt + (Tt - 1)) * Ff + f];
    }
    __syncthreads();

    for (int r = 0; r < Rr; ++r) {
        // ed = relu(xin @ W1 + b1)  (16x2 @ 2x64), written in A-frag layout
        for (int idx = tid; idx < 16 * Ee; idx += 256) {
            int m = idx >> 6, e = idx & 63;
            float v = xin[m * 2 + 0] * W1[e] + xin[m * 2 + 1] * W1[Ee + e] + b1[e];
            v = fmaxf(v, 0.0f);
            edbf[(e >> 5) * 512 + a_slot(m, e & 31)] = f2bf(v);
        }
        __syncthreads();

        v16bf ax0 = load_frag(edbf + lane * 16);
        v16bf ax1 = load_frag(edbf + (32 + lane) * 16);
        lstm_step_reg(ax0, ax1, hbf[pr], hbf[pr ^ 1], c_reg, bbD, WihD, WhhD);
        pr ^= 1;   // this step's h now lives in hbf[pr]

        // out = h @ W2 + b2  (16x128 @ 128x2), feed back as next xin
        if (tid < 32) {
            int m = tid >> 1, f = tid & 1;
            float s = b2[f];
            for (int k = 0; k < Hh; ++k) {
                unsigned short u = hbf[pr][(k >> 5) * 512 + a_slot(m, k & 31)];
                s += bf2f(u) * W2[k * Ff + f];
            }
            out[(((size_t)(b0 + m)) * Rr + r) * Ff + f] = s;
            xin[tid] = s;
        }
        __syncthreads();   // xin ready for next ed
    }
}

// ---------------------------------------------------------------------------
// Host launcher
// ---------------------------------------------------------------------------
extern "C" void kernel_launch(void* const* d_in, const int* in_sizes, int n_in,
                              void* d_out, int out_size, void* d_ws,
                              size_t ws_size, hipStream_t stream) {
    (void)in_sizes; (void)n_in; (void)out_size; (void)ws_size;

    const float* src  = (const float*)d_in[0];
    // d_in[1] = trg (unused by the reference forward pass)
    const float* Wres = (const float*)d_in[2];
    const float* Wg   = (const float*)d_in[3];
    const float* bg   = (const float*)d_in[4];
    const float* WihE = (const float*)d_in[5];
    const float* WhhE = (const float*)d_in[6];
    const float* bE   = (const float*)d_in[7];
    const float* W1   = (const float*)d_in[8];
    const float* b1   = (const float*)d_in[9];
    const float* WihD = (const float*)d_in[10];
    const float* WhhD = (const float*)d_in[11];
    const float* bD   = (const float*)d_in[12];
    const float* W2   = (const float*)d_in[13];
    const float* b2   = (const float*)d_in[14];
    float* out = (float*)d_out;

    // Workspace: bf16-packed encoder inputs, then bf16-packed weights
    char* ws = (char*)d_ws;
    unsigned short* xpk   = (unsigned short*)ws;   // 16*50*2*32*16 = 819200
    unsigned short* pWihE = xpk + (size_t)16 * Tt * 2 * 32 * 16;
    unsigned short* pWhhE = pWihE + 32 * 2 * 32 * 16;   // 32768 elems
    unsigned short* pWihD = pWhhE + 32 * 4 * 32 * 16;   // 65536 elems
    unsigned short* pWhhD = pWihD + 32 * 2 * 32 * 16;

    pack_w_kernel<<<64, 256, 0, stream>>>(WihE, pWihE, 2);
    pack_w_kernel<<<128, 256, 0, stream>>>(WhhE, pWhhE, 4);
    pack_w_kernel<<<64, 256, 0, stream>>>(WihD, pWihD, 2);
    pack_w_kernel<<<128, 256, 0, stream>>>(WhhD, pWhhD, 4);

    stgae_gcn_kernel<<<Bsz * Tt, 64, 0, stream>>>(src, Wres, Wg, bg, xpk);

    stgae_lstm_kernel<<<Bsz / 16, 256, 0, stream>>>(
        xpk, src, pWihE, pWhhE, bE, W1, b1, pWihD, pWhhD, bD, W2, b2, out);
}
